// Attention_20882130993254
// MI455X (gfx1250) — compile-verified, matrix operand order checked by reading, not run on hardware
//
#include <hip/hip_runtime.h>

typedef __bf16 bf16_t;
typedef __attribute__((ext_vector_type(16))) __bf16 v16bf;
typedef __attribute__((ext_vector_type(8)))  __bf16 v8bf;
typedef __attribute__((ext_vector_type(8)))  float  v8f;
typedef __attribute__((ext_vector_type(4)))  unsigned int ui32x4;
typedef __attribute__((ext_vector_type(8)))  int i32x8;
typedef __attribute__((ext_vector_type(4)))  int i32x4;

#define BATCH 16
#define NCTX  2048
#define MQ    512
#define DIM   256
#define ROWS_PER_BLOCK 64
#define THREADS 128
#define NWAVES 4
#define MSTEP 32
#define NSTEPS (MQ / MSTEP)

// ---- Tensor Data Mover: async 2D tile DMA global->LDS, tracked by TENSORcnt.
// ROCm 7.2 builtin is 5-arg; amdgpu-toolchain (therock headers present) is 6-arg.
#if __has_include(<hip/amd_detail/amd_gfx1250_TDM.h>)
#define TDM_LOAD(g0, g1) \
    __builtin_amdgcn_tensor_load_to_lds((g0), (g1), (i32x4)0, (i32x4)0, (i32x8)0, 0)
#else
#define TDM_LOAD(g0, g1) \
    __builtin_amdgcn_tensor_load_to_lds((g0), (g1), (i32x4)0, (i32x4)0, 0)
#endif

// Issue a TDM load of a 32x256 f32 tile (rows contiguous, row stride 256) from
// global `gsrc` into LDS byte offset `lds_off`. D# fields per CDNA5 ISA ch.8:
//   group0: count=1 | lds_addr | global_addr[56:0] | type=2
//   group1: data_size=4B, tensor_dim0=256, tensor_dim1=512,
//           tile_dim0=256, tile_dim1=32, tensor_dim0_stride=256
__device__ __forceinline__ void tdm_load_q_tile(const float* gsrc, unsigned lds_off) {
    unsigned long long ga = (unsigned long long)(size_t)gsrc;
    ui32x4 g0;
    g0[0] = 1u;                                              // count=1, user D#
    g0[1] = lds_off;                                         // lds_addr (bytes)
    g0[2] = (unsigned)(ga & 0xFFFFFFFFu);                    // global_addr[31:0]
    g0[3] = (unsigned)((ga >> 32) & 0x01FFFFFFu) | (2u << 30); // addr[56:32], type=2
    i32x8 g1;
    g1[0] = (int)(2u << 16);                                 // data_size=2 (4 bytes)
    g1[1] = (int)((DIM & 0xFFFFu) << 16);                    // tensor_dim0 lo16 @bit48
    g1[2] = (int)(((DIM >> 16) & 0xFFFF) | ((MQ & 0xFFFF) << 16)); // dim0 hi | dim1 lo
    g1[3] = (int)(((MQ >> 16) & 0xFFFF) | ((unsigned)DIM << 16));  // dim1 hi | tile_dim0
    g1[4] = (int)MSTEP;                                      // tile_dim1=32, tile_dim2=0
    g1[5] = (int)DIM;                                        // tensor_dim0_stride lo32
    g1[6] = 0;                                               // stride hi | dim1_stride lo
    g1[7] = 0;
    TDM_LOAD(g0, g1);
}

// Load a 16x32 bf16 A-style fragment (or B^T-style) from a row-major bf16
// array, following the CDNA5 ISA 16-bit A-matrix VGPR layout:
//   lane L: m = L&15, g = L>>4
//   halves 0..7  -> k = k0 + 8g + 0..7
//   halves 8..15 -> k = k0 + 16 + 8g + 0..7
// Both 8-half runs are contiguous & 16B aligned -> two ds_load_b128.
__device__ __forceinline__ v16bf load_frag_rm(const bf16_t* base, int stride,
                                              int row, int g, int k0) {
    const bf16_t* p = base + row * stride + k0 + 8 * g;
    v8bf lo = *(const v8bf*)(p);
    v8bf hi = *(const v8bf*)(p + 16);
    v16bf f;
#pragma unroll
    for (int i = 0; i < 8; ++i) { f[i] = lo[i]; f[8 + i] = hi[i]; }
    return f;
}

__global__ __launch_bounds__(THREADS)
void bidaf_attention_wmma(const float* __restrict__ Cg,
                          const float* __restrict__ Qg,
                          float* __restrict__ Gg)
{
    __shared__ __align__(16) bf16_t ctile[ROWS_PER_BLOCK][DIM]; // 32 KB context (bf16)
    __shared__ __align__(16) bf16_t qs  [MSTEP][DIM];           // 16 KB Q tile [m][d]
    __shared__ __align__(16) bf16_t qsT [DIM][MSTEP];           // 16 KB Q tile [d][m]
    __shared__ __align__(16) bf16_t pscr[NWAVES][16][MSTEP];    //  4 KB P transpose
    __shared__ __align__(16) float  qf32[2][MSTEP][DIM];        // 64 KB TDM landing pads

    const int b    = blockIdx.y;
    const int n0   = blockIdx.x * ROWS_PER_BLOCK;
    const int tid  = threadIdx.x;
    const int lane = tid & 31;
    const int wave = tid >> 5;
    const int lg   = lane >> 4;   // lane half-group (C/D layout: rows r+8*lg)
    const int ln   = lane & 15;   // column / m within 16

    const float* Cb = Cg + ((size_t)b * NCTX + n0) * DIM;
    const float* Qb = Qg + (size_t)b * MQ * DIM;
    float*       Gb = Gg + ((size_t)b * NCTX + n0) * (2 * DIM);

    // LDS byte offsets of the two TDM landing pads (generic ptr low 32 bits).
    const unsigned qf32_off0 = (unsigned)(size_t)&qf32[0][0][0];
    const unsigned qf32_off1 = (unsigned)(size_t)&qf32[1][0][0];

    // ---- prologue: kick off async DMA of Q tile 0 while we stage C
    if (wave == 0) tdm_load_q_tile(Qb, qf32_off0);

    // ---- Stage context tile -> LDS bf16; fused passthrough C -> G[:, :, 0:DIM]
#pragma unroll 4
    for (int i = tid; i < ROWS_PER_BLOCK * DIM / 4; i += THREADS) {
        float4 v = ((const float4*)Cb)[i];
        int r = i >> 6;          // (i*4)/DIM
        int c = (i & 63) << 2;   // (i*4)%DIM
        ctile[r][c + 0] = (bf16_t)v.x;
        ctile[r][c + 1] = (bf16_t)v.y;
        ctile[r][c + 2] = (bf16_t)v.z;
        ctile[r][c + 3] = (bf16_t)v.w;
        *(float4*)(Gb + (size_t)r * (2 * DIM) + c) = v;
    }

    // ---- flash-attention state: per lane holds rows (r + 8*lg), r = 0..7
    float mrow[8], lrow[8];
    v8f   acc[16];                       // O accumulator: 16 tiles of 16x16 f32
#pragma unroll
    for (int r = 0; r < 8; ++r) { mrow[r] = -3.0e38f; lrow[r] = 0.0f; }
#pragma unroll
    for (int t = 0; t < 16; ++t)
#pragma unroll
        for (int r = 0; r < 8; ++r) acc[t][r] = 0.0f;

    const bf16_t* crow = &ctile[wave * 16][0];
    bf16_t*       ps   = &pscr[wave][0][0];

    for (int it = 0; it < NSTEPS; ++it) {
        const int cur = it & 1;
        // previous iteration's qs/qsT readers done; qf32[cur^1] free for next DMA
        __syncthreads();

        // ---- issue async DMA for tile it+1, then wait for tile it to land.
        // TDM ops from one wave complete in order: with tile it+1 in flight,
        // TENSORcnt<=1 guarantees tile it is fully in LDS.
        if (it + 1 < NSTEPS) {
            if (wave == 0)
                tdm_load_q_tile(Qb + (size_t)(it + 1) * MSTEP * DIM,
                                cur ? qf32_off0 : qf32_off1);
            __builtin_amdgcn_s_wait_tensorcnt(1);
        } else {
            __builtin_amdgcn_s_wait_tensorcnt(0);
        }
        __syncthreads();   // publish qf32[cur] to all waves

        // ---- convert landed f32 tile -> bf16 in both layouts (LDS -> LDS)
#pragma unroll 4
        for (int i = tid; i < MSTEP * DIM / 4; i += THREADS) {
            float4 v = ((const float4*)&qf32[cur][0][0])[i];
            int r = i >> 6;
            int c = (i & 63) << 2;
            bf16_t x = (bf16_t)v.x, y = (bf16_t)v.y, z = (bf16_t)v.z, w = (bf16_t)v.w;
            qs[r][c + 0] = x;  qs[r][c + 1] = y;  qs[r][c + 2] = z;  qs[r][c + 3] = w;
            qsT[c + 0][r] = x; qsT[c + 1][r] = y; qsT[c + 2][r] = z; qsT[c + 3][r] = w;
        }
        __syncthreads();

        // ---- S = C_tile(16x256) x Q_tile^T : two 16x16 f32 tiles, K=256 in 8 steps
        v8f s0, s1;
#pragma unroll
        for (int r = 0; r < 8; ++r) { s0[r] = 0.0f; s1[r] = 0.0f; }
#pragma unroll
        for (int kc = 0; kc < 8; ++kc) {
            v16bf a  = load_frag_rm(crow,      DIM, ln,      lg, kc * 32);  // A: C rows
            v16bf b0 = load_frag_rm(&qs[0][0], DIM, ln,      lg, kc * 32);  // B: Q rows 0..15
            v16bf b1 = load_frag_rm(&qs[0][0], DIM, 16 + ln, lg, kc * 32);  // B: Q rows 16..31
            s0 = __builtin_amdgcn_wmma_f32_16x16x32_bf16(false, a, false, b0,
                                                         (short)0, s0, false, false);
            s1 = __builtin_amdgcn_wmma_f32_16x16x32_bf16(false, a, false, b1,
                                                         (short)0, s1, false, false);
        }

        // ---- online softmax across these 32 columns (row = r + 8*lg per lane)
#pragma unroll
        for (int r = 0; r < 8; ++r) {
            float v = fmaxf(s0[r], s1[r]);
            v = fmaxf(v, __shfl_xor(v, 1, 32));
            v = fmaxf(v, __shfl_xor(v, 2, 32));
            v = fmaxf(v, __shfl_xor(v, 4, 32));
            v = fmaxf(v, __shfl_xor(v, 8, 32));
            float mnew  = fmaxf(mrow[r], v);
            float scale = __expf(mrow[r] - mnew);
            mrow[r] = mnew;
            float p0 = __expf(s0[r] - mnew);
            float p1 = __expf(s1[r] - mnew);
            s0[r] = p0; s1[r] = p1;
            float rs = p0 + p1;
            rs += __shfl_xor(rs, 1, 32);
            rs += __shfl_xor(rs, 2, 32);
            rs += __shfl_xor(rs, 4, 32);
            rs += __shfl_xor(rs, 8, 32);
            lrow[r] = lrow[r] * scale + rs;
#pragma unroll
            for (int t = 0; t < 16; ++t) acc[t][r] *= scale;   // rescale O
        }

        // ---- transpose P: C/D layout -> A layout via per-wave LDS scratch
#pragma unroll
        for (int r = 0; r < 8; ++r) {
            ps[(r + 8 * lg) * MSTEP + ln]      = (bf16_t)s0[r];
            ps[(r + 8 * lg) * MSTEP + 16 + ln] = (bf16_t)s1[r];
        }
        asm volatile("s_wait_dscnt 0" ::: "memory");   // intra-wave LDS RAW fence
        v16bf pa = load_frag_rm(ps, MSTEP, ln, lg, 0);

        // ---- O += P(16x32) x Q_tile(32x256): 16 output tiles, B from qsT
#pragma unroll
        for (int t = 0; t < 16; ++t) {
            v16bf bq = load_frag_rm(&qsT[0][0], MSTEP, t * 16 + ln, lg, 0);
            acc[t] = __builtin_amdgcn_wmma_f32_16x16x32_bf16(false, pa, false, bq,
                                                             (short)0, acc[t], false, false);
        }
    }

    // ---- normalize by softmax denominator and store O -> G[:, :, DIM:2*DIM]
#pragma unroll
    for (int r = 0; r < 8; ++r) {
        float  inv  = 1.0f / lrow[r];
        float* orow = Gb + (size_t)(wave * 16 + r + 8 * lg) * (2 * DIM) + DIM;
#pragma unroll
        for (int t = 0; t < 16; ++t)
            orow[t * 16 + ln] = acc[t][r] * inv;
    }
}

extern "C" void kernel_launch(void* const* d_in, const int* in_sizes, int n_in,
                              void* d_out, int out_size, void* d_ws, size_t ws_size,
                              hipStream_t stream) {
    (void)in_sizes; (void)n_in; (void)out_size; (void)d_ws; (void)ws_size;
    const float* C = (const float*)d_in[0];   // [16, 2048, 256] f32
    const float* Q = (const float*)d_in[1];   // [16,  512, 256] f32
    float*       G = (float*)d_out;           // [16, 2048, 512] f32
    dim3 grid(NCTX / ROWS_PER_BLOCK, BATCH);  // (32, 16)
    bidaf_attention_wmma<<<grid, THREADS, 0, stream>>>(C, Q, G);
}
